// QFormerLayer_81028853006691
// MI455X (gfx1250) — matrix operand order repaired and checked by reading, not compiled, not run on hardware
//
#include <hip/hip_runtime.h>

// ---------------------------------------------------------------------------
// Q-Former block for MI455X (gfx1250, wave32, WMMA f16 16x16x32 w/ f32 accum)
// Double-buffered GEMM main loop with GLOBAL_LOAD_ASYNC_TO_LDS_B128
// (ASYNCcnt-tracked memory->LDS copies, no VGPR staging) when available.
// ---------------------------------------------------------------------------
typedef _Float16 v16h __attribute__((ext_vector_type(16)));
typedef _Float16 v8h  __attribute__((ext_vector_type(8)));
typedef _Float16 v4h  __attribute__((ext_vector_type(4)));
typedef float    v8f  __attribute__((ext_vector_type(8)));
typedef int      v4i  __attribute__((ext_vector_type(4)));

#define DEV __device__ __forceinline__

#if defined(__has_builtin)
#if __has_builtin(__builtin_amdgcn_global_load_async_to_lds_b128)
#define ASYNC_LDS 1
#endif
#endif

DEV void async_copy_b128(const _Float16* g, _Float16* l) {
#ifdef ASYNC_LDS
  // signature (probe-confirmed): (int4 AS(1)* src, int4 AS(3)* dst, imm, imm)
  __builtin_amdgcn_global_load_async_to_lds_b128(
      (__attribute__((address_space(1))) v4i*)g,
      (__attribute__((address_space(3))) v4i*)l, 0, 0);
#else
  (void)g; (void)l;
#endif
}

DEV void wait_async0() {
#ifdef ASYNC_LDS
#if __has_builtin(__builtin_amdgcn_s_wait_asynccnt)
  __builtin_amdgcn_s_wait_asynccnt(0);
#else
  asm volatile("s_wait_asynccnt 0x0" ::: "memory");
#endif
#endif
}

DEV v8f wmma16(v16h a, v16h b, v8f c) {
  // (neg_a, A, neg_b, B, c_mod, C, reuse_a, reuse_b)
  return __builtin_amdgcn_wmma_f32_16x16x32_f16(false, a, false, b, (short)0, c,
                                                false, false);
}

DEV v16h cat16(v8h lo, v8h hi) {
  return __builtin_shufflevector(lo, hi, 0, 1, 2, 3, 4, 5, 6, 7, 8, 9, 10, 11,
                                 12, 13, 14, 15);
}

// A fragment: 16(M) x 32(K) from row-major memory (ld in halves).
// lane&15 = M row; lanes>=16 hold K offsets +8 (ISA 7.12.2 16-bit A layout).
DEV v16h load_afrag(const _Float16* base, int ld, int lane) {
  const _Float16* p = base + (size_t)(lane & 15) * ld + ((lane >> 4) ? 8 : 0);
  v8h lo = *(const v8h*)p;        // VGPR0..3 : K = off .. off+7
  v8h hi = *(const v8h*)(p + 16); // VGPR4..7 : K = 16+off .. 16+off+7
  return cat16(lo, hi);
}

// B fragment: 32(K) x 16(N); source is row-major [N][K] (we compute A*B^T).
// lane&15 = N row; lanes>=16 hold K offsets +8 (same K order as A).
DEV v16h load_bfrag(const _Float16* base, int ld, int lane) {
  const _Float16* p = base + (size_t)(lane & 15) * ld + ((lane >> 4) ? 8 : 0);
  v8h lo = *(const v8h*)p;
  v8h hi = *(const v8h*)(p + 16);
  return cat16(lo, hi);
}

// ---------------------------------------------------------------------------
// Tiled WMMA GEMM:  C[M,N] = A[M,K] * B[N,K]^T   (both f16 row-major, K-contig)
// Block tile 128x128, k-step 32, 8 waves (4M x 2N), wave tile 32x64.
// Double-buffered LDS stages; async memory->LDS copies when available.
// ---------------------------------------------------------------------------
enum { MODE_F16 = 0, MODE_F32 = 1, MODE_F32RES = 2, MODE_F32RES_F16 = 3,
       MODE_KV = 4 };

template <int MODE>
__global__ __launch_bounds__(256) void gemm_wmma_k(
    const _Float16* __restrict__ A, const _Float16* __restrict__ B, int M,
    int N, int K, float* __restrict__ Cf, const float* __restrict__ Res,
    _Float16* __restrict__ C16) {
  __shared__ _Float16 Ash[2][128 * 40];  // 80B row stride: 16B-aligned frags
  __shared__ _Float16 Bsh[2][128 * 40];
  const int t = threadIdx.x, lane = t & 31, w = t >> 5;
  const int mBase = blockIdx.y * 128, nBase = blockIdx.x * 128;
  const int wm = (w >> 1) * 32, wn = (w & 1) * 64;
  const int lrow = t >> 1, lcol = (t & 1) * 16;

  const _Float16* gA = A + (size_t)(mBase + lrow) * K + lcol;
  const _Float16* gB = B + (size_t)(nBase + lrow) * K + lcol;
  _Float16* lA0 = &Ash[0][lrow * 40 + lcol];
  _Float16* lB0 = &Bsh[0][lrow * 40 + lcol];
  const int lbuf = 128 * 40;  // halves between stage 0 and stage 1

  v8f acc[2][4];
#pragma unroll
  for (int mi = 0; mi < 2; mi++)
#pragma unroll
    for (int ni = 0; ni < 4; ni++)
#pragma unroll
      for (int r = 0; r < 8; r++) acc[mi][ni][r] = 0.f;

#ifdef ASYNC_LDS
  // ---- async pipeline: issue -> (compute prev) -> s_wait_asynccnt -> barrier
  {
    async_copy_b128(gA, lA0);
    async_copy_b128(gA + 8, lA0 + 8);
    async_copy_b128(gB, lB0);
    async_copy_b128(gB + 8, lB0 + 8);
  }
  wait_async0();
  __syncthreads();
  for (int k0 = 0; k0 < K; k0 += 32) {
    const int cur = (k0 >> 5) & 1, nxt = cur ^ 1;
    if (k0 + 32 < K) {
      const _Float16* ga = gA + k0 + 32;
      const _Float16* gb = gB + k0 + 32;
      async_copy_b128(ga, lA0 + nxt * lbuf);
      async_copy_b128(ga + 8, lA0 + nxt * lbuf + 8);
      async_copy_b128(gb, lB0 + nxt * lbuf);
      async_copy_b128(gb + 8, lB0 + nxt * lbuf + 8);
    }
    v16h af[2], bf[4];
#pragma unroll
    for (int mi = 0; mi < 2; mi++)
      af[mi] = load_afrag(&Ash[cur][(wm + mi * 16) * 40], 40, lane);
#pragma unroll
    for (int ni = 0; ni < 4; ni++)
      bf[ni] = load_bfrag(&Bsh[cur][(wn + ni * 16) * 40], 40, lane);
#pragma unroll
    for (int mi = 0; mi < 2; mi++)
#pragma unroll
      for (int ni = 0; ni < 4; ni++)
        acc[mi][ni] = wmma16(af[mi], bf[ni], acc[mi][ni]);
    wait_async0();
    __syncthreads();
  }
#else
  // ---- fallback: VGPR-staged double buffer, one barrier per k-step
  uint4 a0, a1, b0, b1;
  {
    const uint4* pa = (const uint4*)gA;
    const uint4* pb = (const uint4*)gB;
    a0 = pa[0]; a1 = pa[1]; b0 = pb[0]; b1 = pb[1];
    *(uint4*)lA0 = a0; *(uint4*)(lA0 + 8) = a1;
    *(uint4*)lB0 = b0; *(uint4*)(lB0 + 8) = b1;
  }
  __syncthreads();
  for (int k0 = 0; k0 < K; k0 += 32) {
    const int cur = (k0 >> 5) & 1, nxt = cur ^ 1;
    if (k0 + 32 < K) {
      const uint4* pa = (const uint4*)(gA + k0 + 32);
      const uint4* pb = (const uint4*)(gB + k0 + 32);
      a0 = pa[0]; a1 = pa[1]; b0 = pb[0]; b1 = pb[1];
    }
    v16h af[2], bf[4];
#pragma unroll
    for (int mi = 0; mi < 2; mi++)
      af[mi] = load_afrag(&Ash[cur][(wm + mi * 16) * 40], 40, lane);
#pragma unroll
    for (int ni = 0; ni < 4; ni++)
      bf[ni] = load_bfrag(&Bsh[cur][(wn + ni * 16) * 40], 40, lane);
#pragma unroll
    for (int mi = 0; mi < 2; mi++)
#pragma unroll
      for (int ni = 0; ni < 4; ni++)
        acc[mi][ni] = wmma16(af[mi], bf[ni], acc[mi][ni]);
    if (k0 + 32 < K) {
      *(uint4*)(lA0 + nxt * lbuf) = a0;
      *(uint4*)(lA0 + nxt * lbuf + 8) = a1;
      *(uint4*)(lB0 + nxt * lbuf) = b0;
      *(uint4*)(lB0 + nxt * lbuf + 8) = b1;
    }
    __syncthreads();
  }
#endif

  const int rowHalf = (lane >> 4) ? 8 : 0;  // C/D layout (ISA 7.12.2)
  const int colL = lane & 15;
#pragma unroll
  for (int mi = 0; mi < 2; mi++)
#pragma unroll
    for (int ni = 0; ni < 4; ni++)
#pragma unroll
      for (int r = 0; r < 8; r++) {
        int row = mBase + wm + mi * 16 + rowHalf + r;
        int col = nBase + wn + ni * 16 + colL;
        float v = acc[mi][ni][r];
        size_t idx = (size_t)row * N + col;
        if (MODE == MODE_F16) {
          C16[idx] = (_Float16)v;
        } else if (MODE == MODE_F32) {
          Cf[idx] = v;
        } else if (MODE == MODE_F32RES) {
          Cf[idx] = Res[idx] + v;
        } else if (MODE == MODE_F32RES_F16) {
          float x = Res[idx] + v;
          Cf[idx] = x;
          C16[idx] = (_Float16)x;
        } else {  // MODE_KV: col = l*1024 + kvh*128 + kvd ; row = b*256 + q
          int l = col >> 10, rem = col & 1023, kvh = rem >> 7, kvd = rem & 127;
          int b = row >> 8, q = row & 255;
          size_t o = ((((size_t)l * 4 + b) * 8 + kvh) * 256 + q) * 128 + kvd;
          Cf[o] = v;
        }
      }
}

// ---------------------------------------------------------------------------
// LayerNorm: one wave per 1024-elem row, f32 in -> f16 out.
// ---------------------------------------------------------------------------
__global__ __launch_bounds__(256) void ln_k(const float* __restrict__ X,
                                            const float* __restrict__ G,
                                            const float* __restrict__ Bb,
                                            _Float16* __restrict__ Y) {
  int row = blockIdx.x * 8 + (threadIdx.x >> 5);
  int lane = threadIdx.x & 31;
  const float4* x4 = (const float4*)(X + (size_t)row * 1024);
  float4 v[8];
  float s = 0.f, ss = 0.f;
#pragma unroll
  for (int i = 0; i < 8; i++) {
    v[i] = x4[lane + i * 32];
    s += v[i].x + v[i].y + v[i].z + v[i].w;
    ss += v[i].x * v[i].x + v[i].y * v[i].y + v[i].z * v[i].z + v[i].w * v[i].w;
  }
#pragma unroll
  for (int m = 1; m < 32; m <<= 1) {
    s += __shfl_xor(s, m, 32);
    ss += __shfl_xor(ss, m, 32);
  }
  float mean = s * (1.f / 1024.f);
  float var = ss * (1.f / 1024.f) - mean * mean;
  float inv = rsqrtf(var + 1e-5f);
  _Float16* y = Y + (size_t)row * 1024;
#pragma unroll
  for (int i = 0; i < 8; i++) {
    int j = (lane + i * 32) * 4;
    v4h o;
    o[0] = (_Float16)((v[i].x - mean) * inv * G[j + 0] + Bb[j + 0]);
    o[1] = (_Float16)((v[i].y - mean) * inv * G[j + 1] + Bb[j + 1]);
    o[2] = (_Float16)((v[i].z - mean) * inv * G[j + 2] + Bb[j + 2]);
    o[3] = (_Float16)((v[i].w - mean) * inv * G[j + 3] + Bb[j + 3]);
    *(v4h*)(y + j) = o;
  }
}

// ---------------------------------------------------------------------------
// Elementwise kernels
// ---------------------------------------------------------------------------
__global__ void cvt_f32_f16_k(const float* __restrict__ in,
                              _Float16* __restrict__ out, int n4) {
  int i = blockIdx.x * 256 + threadIdx.x;
  if (i < n4) {
    float4 a = ((const float4*)in)[i];
    v4h o;
    o[0] = (_Float16)a.x; o[1] = (_Float16)a.y;
    o[2] = (_Float16)a.z; o[3] = (_Float16)a.w;
    ((v4h*)out)[i] = o;
  }
}

__global__ void silu_mul_k(const float* __restrict__ g,
                           const float* __restrict__ u,
                           _Float16* __restrict__ out, int n4) {
  int i = blockIdx.x * 256 + threadIdx.x;
  if (i < n4) {
    float4 a = ((const float4*)g)[i];
    float4 b = ((const float4*)u)[i];
    v4h o;
    o[0] = (_Float16)(a.x / (1.f + __expf(-a.x)) * b.x);
    o[1] = (_Float16)(a.y / (1.f + __expf(-a.y)) * b.y);
    o[2] = (_Float16)(a.z / (1.f + __expf(-a.z)) * b.z);
    o[3] = (_Float16)(a.w / (1.f + __expf(-a.w)) * b.w);
    ((v4h*)out)[i] = o;
  }
}

// ---------------------------------------------------------------------------
// Dense self-attention, one block per (b, head): seq 256, head dim 64.
// WMMA scores (C-frags in regs), shfl softmax, P via per-wave LDS tile, P*V^T.
// ---------------------------------------------------------------------------
__global__ __launch_bounds__(256) void self_attn_k(
    const _Float16* __restrict__ Q, const _Float16* __restrict__ Km,
    const _Float16* __restrict__ V, _Float16* __restrict__ O) {
  __shared__ _Float16 vT[64 * 264];      // V^T [dim][seq], 528B row stride
  __shared__ _Float16 Pt[8 * 16 * 40];   // per-wave P tile [16][32] (+pad)
  const int b = blockIdx.x >> 4, h = blockIdx.x & 15;
  const int t = threadIdx.x, lane = t & 31, w = t >> 5;

  {  // stage V transposed: thread t owns seq index t
    const _Float16* vp = V + (size_t)(b * 256 + t) * 1024 + h * 64;
#pragma unroll
    for (int i = 0; i < 8; i++) {
      v8h r = *(const v8h*)(vp + i * 8);
#pragma unroll
      for (int c = 0; c < 8; c++) vT[(i * 8 + c) * 264 + t] = r[c];
    }
  }
  __syncthreads();

  const int hi = lane >> 4;
  const int rowHalf = hi ? 8 : 0;
  _Float16* myP = Pt + w * (16 * 40);

  for (int chunk = 0; chunk < 2; chunk++) {
    int m0 = w * 32 + chunk * 16;
    // Q fragments straight from global (head dim is K-contiguous), pre-scaled
    v16h qf[2];
    {
      const _Float16* qp =
          Q + (size_t)(b * 256 + m0 + (lane & 15)) * 1024 + h * 64;
      int off = hi ? 8 : 0;
#pragma unroll
      for (int ks = 0; ks < 2; ks++) {
        v8h lo = *(const v8h*)(qp + ks * 32 + off);
        v8h hv = *(const v8h*)(qp + ks * 32 + 16 + off);
        v16h f = cat16(lo, hv);
#pragma unroll
        for (int e = 0; e < 16; e++) f[e] = f[e] * (_Float16)0.125f;
        qf[ks] = f;
      }
    }
    // scores S[16 rows][256 cols] as 16 C-fragments
    v8f S[16];
#pragma unroll
    for (int j = 0; j < 16; j++) {
#pragma unroll
      for (int r = 0; r < 8; r++) S[j][r] = 0.f;
      const _Float16* kp =
          Km + (size_t)(b * 256 + j * 16 + (lane & 15)) * 1024 + h * 64;
      int koff = hi ? 8 : 0;
#pragma unroll
      for (int ks = 0; ks < 2; ks++) {
        v8h lo = *(const v8h*)(kp + ks * 32 + koff);
        v8h hv = *(const v8h*)(kp + ks * 32 + koff + 16);
        S[j] = wmma16(qf[ks], cat16(lo, hv), S[j]);
      }
    }
    // softmax: reduce across N (lanes within each 16-lane half)
    float rm[8], se[8];
#pragma unroll
    for (int r = 0; r < 8; r++) {
      float m = -1e30f;
#pragma unroll
      for (int j = 0; j < 16; j++) m = fmaxf(m, S[j][r]);
#pragma unroll
      for (int x = 1; x < 16; x <<= 1) m = fmaxf(m, __shfl_xor(m, x, 32));
      rm[r] = m;
    }
#pragma unroll
    for (int r = 0; r < 8; r++) {
      float sum = 0.f;
#pragma unroll
      for (int j = 0; j < 16; j++) {
        float pv = __expf(S[j][r] - rm[r]);
        S[j][r] = pv;
        sum += pv;
      }
#pragma unroll
      for (int x = 1; x < 16; x <<= 1) sum += __shfl_xor(sum, x, 32);
      se[r] = 1.f / sum;
    }
    // O = P * V^T via WMMA; P goes C-layout -> LDS -> A-layout
    v8f Ofr[4];
#pragma unroll
    for (int nd = 0; nd < 4; nd++)
#pragma unroll
      for (int r = 0; r < 8; r++) Ofr[nd][r] = 0.f;
#pragma unroll
    for (int kb = 0; kb < 8; kb++) {
#pragma unroll
      for (int jj = 0; jj < 2; jj++) {
        int j = kb * 2 + jj;
#pragma unroll
        for (int r = 0; r < 8; r++)
          myP[(rowHalf + r) * 40 + jj * 16 + (lane & 15)] =
              (_Float16)(S[j][r] * se[r]);
      }
      // wave-local LDS write->read; DS in-order per wave, wait for safety
      asm volatile("s_wait_dscnt 0" ::: "memory");
      v16h af = load_afrag(myP, 40, lane);
      int koff = hi ? 8 : 0;
#pragma unroll
      for (int nd = 0; nd < 4; nd++) {
        const _Float16* vp = &vT[(nd * 16 + (lane & 15)) * 264 + kb * 32 + koff];
        v8h lo = *(const v8h*)vp;
        v8h hv = *(const v8h*)(vp + 16);
        Ofr[nd] = wmma16(af, cat16(lo, hv), Ofr[nd]);
      }
    }
#pragma unroll
    for (int nd = 0; nd < 4; nd++)
#pragma unroll
      for (int r = 0; r < 8; r++)
        O[(size_t)(b * 256 + m0 + rowHalf + r) * 1024 + h * 64 + nd * 16 +
          (lane & 15)] = (_Float16)Ofr[nd][r];
  }
}

// ---------------------------------------------------------------------------
// Windowed cross-attention: each query attends kv in [8q, 8q+8] per doc
// (<=36 positions of 8192): sparse VALU path, one wave per (b,h,q).
// ---------------------------------------------------------------------------
__global__ __launch_bounds__(256) void cross_attn_k(
    const _Float16* __restrict__ Q, const _Float16* __restrict__ Kc,
    const _Float16* __restrict__ Vc, _Float16* __restrict__ O) {
  int gw = blockIdx.x * 8 + (threadIdx.x >> 5);
  int lane = threadIdx.x & 31;
  int q = gw & 255, h = (gw >> 8) & 15, b = gw >> 12;
  const _Float16* qp = Q + (size_t)(b * 256 + q) * 1024 + h * 64 + lane * 2;
  float q0 = (float)qp[0] * 0.125f, q1 = (float)qp[1] * 0.125f;
  int kv0 = q * 8;
  float s[36];
#pragma unroll
  for (int doc = 0; doc < 4; doc++)
#pragma unroll
    for (int j = 0; j < 9; j++) {
      float d = -1e30f;
      if (kv0 + j < 2048) {
        const _Float16* kp = Kc +
            ((size_t)b * 8192 + doc * 2048 + kv0 + j) * 1024 + h * 64 + lane * 2;
        d = q0 * (float)kp[0] + q1 * (float)kp[1];
#pragma unroll
        for (int x = 1; x < 32; x <<= 1) d += __shfl_xor(d, x, 32);
      }
      s[doc * 9 + j] = d;
    }
  float mx = -1e30f;
#pragma unroll
  for (int i = 0; i < 36; i++) mx = fmaxf(mx, s[i]);
  float sum = 0.f;
#pragma unroll
  for (int i = 0; i < 36; i++) {
    s[i] = __expf(s[i] - mx);
    sum += s[i];
  }
  float inv = 1.f / sum;
  float o0 = 0.f, o1 = 0.f;
#pragma unroll
  for (int doc = 0; doc < 4; doc++)
#pragma unroll
    for (int j = 0; j < 9; j++)
      if (kv0 + j < 2048) {
        const _Float16* vp = Vc +
            ((size_t)b * 8192 + doc * 2048 + kv0 + j) * 1024 + h * 64 + lane * 2;
        float pw = s[doc * 9 + j] * inv;
        o0 += pw * (float)vp[0];
        o1 += pw * (float)vp[1];
      }
  _Float16* op = O + (size_t)(b * 256 + q) * 1024 + h * 64 + lane * 2;
  op[0] = (_Float16)o0;
  op[1] = (_Float16)o1;
}

// ---------------------------------------------------------------------------
// Host orchestration
// ---------------------------------------------------------------------------
extern "C" void kernel_launch(void* const* d_in, const int* in_sizes, int n_in,
                              void* d_out, int out_size, void* d_ws,
                              size_t ws_size, hipStream_t stream) {
  (void)in_sizes; (void)n_in; (void)out_size; (void)ws_size;
  const float* query = (const float*)d_in[0];
  const float* hidden = (const float*)d_in[1];
  const float* ln1_g = (const float*)d_in[2];
  const float* ln1_b = (const float*)d_in[3];
  const float* ln2_g = (const float*)d_in[4];
  const float* ln2_b = (const float*)d_in[5];
  const float* lnkv_g = (const float*)d_in[6];
  const float* lnkv_b = (const float*)d_in[7];
  const float* lnf_g = (const float*)d_in[8];
  const float* lnf_b = (const float*)d_in[9];

  char* p = (char*)d_ws;
  auto alloc = [&](size_t bytes) -> char* {
    char* r = p;
    p += (bytes + 255) & ~(size_t)255;
    return r;
  };
  const size_t M1 = 1024;   // B*NQ rows
  const size_t MH = 32768;  // B*KVLEN rows
  const size_t H = 1024;

  // f16 weight copies (indices into d_in: 10..22)
  const size_t wElems[13] = {H * H, H * H, H * H, H * H, H * H, H * H, H * H,
                             H * H, 4096 * H, 4096 * H, H * 4096,
                             4 * 1024 * H, 4 * 1024 * H};
  _Float16* wh[13];
  for (int i = 0; i < 13; i++) {
    wh[i] = (_Float16*)alloc(wElems[i] * 2);
    int n4 = (int)(wElems[i] / 4);
    cvt_f32_f16_k<<<(n4 + 255) / 256, 256, 0, stream>>>(
        (const float*)d_in[10 + i], wh[i], n4);
  }
  _Float16 *wqs = wh[0], *wks = wh[1], *wvs = wh[2], *wos = wh[3];
  _Float16 *wqc = wh[4], *wkc = wh[5], *wvc = wh[6], *woc = wh[7];
  _Float16 *w1h = wh[8], *w2h = wh[9], *w3h = wh[10];
  _Float16 *wokh = wh[11], *wovh = wh[12];

  _Float16* xln = (_Float16*)alloc(M1 * H * 2);
  _Float16* qs = (_Float16*)alloc(M1 * H * 2);
  _Float16* ks = (_Float16*)alloc(M1 * H * 2);
  _Float16* vs = (_Float16*)alloc(M1 * H * 2);
  _Float16* attn_s = (_Float16*)alloc(M1 * H * 2);
  _Float16* hsln = (_Float16*)alloc(MH * H * 2);
  _Float16* kc = (_Float16*)alloc(MH * H * 2);
  _Float16* vc = (_Float16*)alloc(MH * H * 2);
  _Float16* qc = (_Float16*)alloc(M1 * H * 2);
  _Float16* attn_c = (_Float16*)alloc(M1 * H * 2);
  _Float16* hffn = (_Float16*)alloc(M1 * 4096 * 2);
  _Float16* qt3h = (_Float16*)alloc(M1 * H * 2);
  float* qt1 = (float*)alloc(M1 * H * 4);
  float* qt2 = (float*)alloc(M1 * H * 4);
  float* gbuf = (float*)alloc(M1 * 4096 * 4);
  float* ubuf = (float*)alloc(M1 * 4096 * 4);

  float* out_qt = (float*)d_out;                    // [B,NQ,H]
  float* out_k = out_qt + M1 * H;                   // [L,B,KVH,NQ,KVD]
  float* out_v = out_k + 4ull * 4 * 8 * 256 * 128;  // same shape

  auto gemm = [&](int mode, const _Float16* A, const _Float16* Bm, int M,
                  int N, int K, float* Cf, const float* Res, _Float16* C16) {
    dim3 grid(N / 128, M / 128);
    switch (mode) {
      case MODE_F16:
        gemm_wmma_k<MODE_F16><<<grid, 256, 0, stream>>>(A, Bm, M, N, K, Cf, Res, C16);
        break;
      case MODE_F32:
        gemm_wmma_k<MODE_F32><<<grid, 256, 0, stream>>>(A, Bm, M, N, K, Cf, Res, C16);
        break;
      case MODE_F32RES:
        gemm_wmma_k<MODE_F32RES><<<grid, 256, 0, stream>>>(A, Bm, M, N, K, Cf, Res, C16);
        break;
      case MODE_F32RES_F16:
        gemm_wmma_k<MODE_F32RES_F16><<<grid, 256, 0, stream>>>(A, Bm, M, N, K, Cf, Res, C16);
        break;
      default:
        gemm_wmma_k<MODE_KV><<<grid, 256, 0, stream>>>(A, Bm, M, N, K, Cf, Res, C16);
        break;
    }
  };

  // 1) self-attention
  ln_k<<<128, 256, 0, stream>>>(query, ln1_g, ln1_b, xln);
  gemm(MODE_F16, xln, wqs, 1024, 1024, 1024, nullptr, nullptr, qs);
  gemm(MODE_F16, xln, wks, 1024, 1024, 1024, nullptr, nullptr, ks);
  gemm(MODE_F16, xln, wvs, 1024, 1024, 1024, nullptr, nullptr, vs);
  self_attn_k<<<64, 256, 0, stream>>>(qs, ks, vs, attn_s);
  gemm(MODE_F32RES, attn_s, wos, 1024, 1024, 1024, qt1, query, nullptr);

  // 2) windowed cross-attention
  ln_k<<<128, 256, 0, stream>>>(qt1, ln2_g, ln2_b, xln);
  gemm(MODE_F16, xln, wqc, 1024, 1024, 1024, nullptr, nullptr, qc);
  ln_k<<<4096, 256, 0, stream>>>(hidden, lnkv_g, lnkv_b, hsln);
  gemm(MODE_F16, hsln, wkc, 32768, 1024, 1024, nullptr, nullptr, kc);
  gemm(MODE_F16, hsln, wvc, 32768, 1024, 1024, nullptr, nullptr, vc);
  cross_attn_k<<<2048, 256, 0, stream>>>(qc, kc, vc, attn_c);
  gemm(MODE_F32RES, attn_c, woc, 1024, 1024, 1024, qt2, qt1, nullptr);

  // 3) SwiGLU FFN
  ln_k<<<128, 256, 0, stream>>>(qt2, lnf_g, lnf_b, xln);
  gemm(MODE_F32, xln, w1h, 1024, 4096, 1024, gbuf, nullptr, nullptr);
  gemm(MODE_F32, xln, w2h, 1024, 4096, 1024, ubuf, nullptr, nullptr);
  silu_mul_k<<<(1024 * 1024), 256, 0, stream>>>(gbuf, ubuf, hffn,
                                                1024 * 4096 / 4);
  gemm(MODE_F32RES_F16, hffn, w3h, 1024, 1024, 4096, out_qt, qt2, qt3h);

  // 4) per-layer K/V projections with fused [L,B,KVH,NQ,KVD] scatter
  gemm(MODE_KV, qt3h, wokh, 1024, 4096, 1024, out_k, nullptr, nullptr);
  gemm(MODE_KV, qt3h, wovh, 1024, 4096, 1024, out_v, nullptr, nullptr);
}
